// GumbelVectorQuantizer_76493367542079
// MI455X (gfx1250) — matrix operand order repaired and checked by reading, not compiled, not run on hardware
//
#include <hip/hip_runtime.h>

// ---------------------------------------------------------------------------
// Gumbel Vector Quantizer forward, fused for MI455X (gfx1250, wave32, WMMA).
//
// logits = x[16384,1024] @ W^T[1024,640]  (~21.5 GFLOP) via bf16 WMMA with
// f32 accumulate: keeps the GEMM under the HBM roofline (~143MB mandatory
// traffic ~ 6us @ 23.3 TB/s); f32 WMMA would be compute bound.
//   - x read once from HBM, converted f32->bf16 in registers (v_cvt_pk_bf16)
//   - W pre-converted to bf16 (1.3MB, L2 resident), staged per-32K chunk into
//     LDS by the Tensor Data Mover (tensor_load_to_lds), double-buffered so
//     the DMA for chunk i+1 overlaps WMMA on chunk i (s_wait_tensorcnt).
//   - sched_group_barrier hints pipeline the inner loop: 3 B fragments of
//     ds reads in flight ahead of each WMMA (no s_wait_dscnt 0 per WMMA).
//   - softmax stats / argmax / codebook gather fused; logits never hit HBM.
// ---------------------------------------------------------------------------

typedef __attribute__((ext_vector_type(16))) __bf16        v16bf;
typedef __attribute__((ext_vector_type(8)))  float         v8f;
typedef __attribute__((ext_vector_type(4)))  unsigned int  v4u;
typedef __attribute__((ext_vector_type(8)))  int           v8i;
typedef __attribute__((ext_vector_type(4)))  int           v4i;

#define NROWS   16384     // B*T
#define C_DIM   1024
#define NGROUPS 2
#define VCODES  320
#define GVDIM   640       // NGROUPS*VCODES
#define D_DIM   256
#define KSTEP   32        // WMMA K per step (bf16)
#define BLK_M   64        // rows per block
#define TILES_N 20        // 320/16 WMMA N-tiles per wave
#define CHUNK_ELE (GVDIM * KSTEP)   // 20480 bf16 = 40KB per k-chunk

#if defined(__has_builtin)
#if __has_builtin(__builtin_amdgcn_tensor_load_to_lds)
#define GVQ_TDM 1
#endif
#if __has_builtin(__builtin_amdgcn_sched_group_barrier)
#define GVQ_SCHED 1
#endif
#endif
#ifndef GVQ_TDM
#define GVQ_TDM 0
#endif
#ifndef GVQ_SCHED
#define GVQ_SCHED 0
#endif

// ---------------------------------------------------------------------------
__global__ void gvq_zero_kernel(float* __restrict__ p, int n) {
  int i = blockIdx.x * blockDim.x + threadIdx.x;
  if (i < n) p[i] = 0.0f;
}

// W f32 -> bf16 (RNE via hardware cvt)
__global__ void gvq_wconv_kernel(const float* __restrict__ W,
                                 unsigned short* __restrict__ Wb, int n) {
  int i = blockIdx.x * blockDim.x + threadIdx.x;
  if (i < n) {
    __bf16 v = (__bf16)W[i];
    Wb[i] = __builtin_bit_cast(unsigned short, v);
  }
}

// ---------------------------------------------------------------------------
// Stage one 640x32 bf16 W-chunk into LDS.
//  TDM path: 2D tensor descriptor, elements are 2B; tile rows pack
//  consecutively into LDS => exactly Bs[col][k] row-major layout.
// ---------------------------------------------------------------------------
__device__ __forceinline__ void gvq_stage_chunk(unsigned short* dstLds,
                                                const unsigned short* src,
                                                int tid, int wave) {
#if GVQ_TDM
  if (wave == 0) {
    const unsigned int lds_off =
        (unsigned int)(unsigned long long)(__SIZE_TYPE__)dstLds;   // low 32b = LDS offset
    const unsigned long long ga = (unsigned long long)(__SIZE_TYPE__)src;
    v4u g0;
    g0[0] = 1u;                                   // count=1, user descriptor
    g0[1] = lds_off;                              // lds_addr
    g0[2] = (unsigned int)(ga & 0xFFFFFFFFu);     // global_addr[31:0]
    g0[3] = (unsigned int)((ga >> 32) & 0x01FFFFFFu) | (2u << 30); // addr hi | type=2
    v8i g1;
    g1[0] = 0x00010000;                           // wg_mask=0, data_size=1 (2B)
    g1[1] = (int)((unsigned)C_DIM << 16);         // tensor_dim0[15:0] @ bits 63:48
    g1[2] = (int)((unsigned)GVDIM << 16);         // tensor_dim1[15:0] @ bits 95:80
    g1[3] = (int)((unsigned)KSTEP << 16);         // tile_dim0 @ bits 127:112
    g1[4] = GVDIM;                                // tile_dim1 @ bits 143:128
    g1[5] = C_DIM;                                // tensor_dim0_stride[31:0]
    g1[6] = 0;
    g1[7] = 0;
    v4i g2; g2[0] = 0; g2[1] = 0; g2[2] = 0; g2[3] = 0;   // 2D: groups 2/3 unused
    v4i g3; g3[0] = 0; g3[1] = 0; g3[2] = 0; g3[3] = 0;
#if defined(__clang_major__) && (__clang_major__ >= 23)
    v8i g4; g4[0]=0; g4[1]=0; g4[2]=0; g4[3]=0; g4[4]=0; g4[5]=0; g4[6]=0; g4[7]=0;
    __builtin_amdgcn_tensor_load_to_lds(g0, g1, g2, g3, g4, 0);
#else
    __builtin_amdgcn_tensor_load_to_lds(g0, g1, g2, g3, 0);
#endif
  }
#else
  // fallback: cooperative 16B-chunk copy (2560 chunks, 10 per thread)
  #pragma unroll
  for (int q = tid; q < CHUNK_ELE / 8; q += 256) {
    const int c = q >> 2, part = q & 3;
    *(uint4*)&dstLds[c * KSTEP + part * 8] =
        *(const uint4*)&src[c * C_DIM + part * 8];
  }
#endif
}

__device__ __forceinline__ void gvq_stage_wait(int wave) {
#if GVQ_TDM
  if (wave == 0) __builtin_amdgcn_s_wait_tensorcnt(0);
#endif
}

// ---------------------------------------------------------------------------
// Fused GEMM (bf16 WMMA) + softmax stats + argmax + codebook gather.
// 256 threads = 8 waves: (mw 0..3) x (gw 0..1); wave owns 16 rows x one
// full group of 320 codes -> complete softmax row in registers.
// ---------------------------------------------------------------------------
__launch_bounds__(256)
__global__ void gvq_fused_kernel(const float* __restrict__ x,
                                 const unsigned short* __restrict__ Wb,
                                 const float* __restrict__ bias,
                                 const float* __restrict__ codebook,
                                 const float* __restrict__ gn,
                                 float* __restrict__ out,
                                 float* __restrict__ g_probs,
                                 float* __restrict__ g_counts) {
  // 80KB: double-buffered W chunk; stats overlay the buffers (only live
  // after the k-loop, when Bs is dead).
  __shared__ __attribute__((aligned(64))) union {
    unsigned short Bs[2][CHUNK_ELE];
    struct {
      float probs[GVDIM];
      float counts[GVDIM];
      int   idx[NGROUPS * BLK_M];
    } st;
  } smem;

  const int tid  = threadIdx.x;
  const int wave = tid >> 5;
  const int lane = tid & 31;
  const int mw   = wave & 3;     // M sub-tile 0..3
  const int gw   = wave >> 2;    // group (N half) 0..1
  const int ln   = lane & 15;
  const int h    = lane >> 4;
  const int blk  = blockIdx.x;
  const int m0   = blk * BLK_M + mw * 16;

  v8f acc[TILES_N];
  #pragma unroll
  for (int t = 0; t < TILES_N; ++t)
    #pragma unroll
    for (int j = 0; j < 8; ++j) acc[t][j] = 0.0f;

  // A fragment rows: lanes 0-15 and 16-31 both hold M = ln (K split by h)
  const float* xr = x + (size_t)(m0 + ln) * C_DIM;

  // prologue: stage chunk 0 into buffer 0
  gvq_stage_chunk(smem.Bs[0], Wb, tid, wave);
  gvq_stage_wait(wave);
  __syncthreads();

  for (int i = 0; i < C_DIM / KSTEP; ++i) {
    const int kb  = i * KSTEP;
    const int cur = i & 1;
    // kick DMA for next chunk into the other buffer (readers of that buffer
    // finished at the barrier ending iteration i-1)
    if (kb + KSTEP < C_DIM)
      gvq_stage_chunk(smem.Bs[cur ^ 1], Wb + (kb + KSTEP), tid, wave);

    // Build A fragment (16x32 bf16) from global f32 x.
    // lane half h: elems 0-15 = K kb+16h..kb+16h+15
    float4 a0 = *(const float4*)&xr[kb + h * 16];
    float4 a1 = *(const float4*)&xr[kb + h * 16 + 4];
    float4 a2 = *(const float4*)&xr[kb + h * 16 + 8];
    float4 a3 = *(const float4*)&xr[kb + h * 16 + 12];
    v16bf a;
    a[0]  = (__bf16)a0.x; a[1]  = (__bf16)a0.y; a[2]  = (__bf16)a0.z; a[3]  = (__bf16)a0.w;
    a[4]  = (__bf16)a1.x; a[5]  = (__bf16)a1.y; a[6]  = (__bf16)a1.z; a[7]  = (__bf16)a1.w;
    a[8]  = (__bf16)a2.x; a[9]  = (__bf16)a2.y; a[10] = (__bf16)a2.z; a[11] = (__bf16)a2.w;
    a[12] = (__bf16)a3.x; a[13] = (__bf16)a3.y; a[14] = (__bf16)a3.z; a[15] = (__bf16)a3.w;

    // 20 WMMAs with a 4-slot rotating B register file.
    const unsigned short* bbase = smem.Bs[cur] + (gw * VCODES + ln) * KSTEP + h * 16;
    #define GVQ_LOADB(t) (*(const v16bf*)&bbase[(t) * 16 * KSTEP])
    v16bf b0 = GVQ_LOADB(0);
    v16bf b1 = GVQ_LOADB(1);
    v16bf b2 = GVQ_LOADB(2);
    v16bf b3 = GVQ_LOADB(3);
    #pragma unroll
    for (int t = 0; t < TILES_N; t += 4) {
      acc[t + 0] = __builtin_amdgcn_wmma_f32_16x16x32_bf16(
          false, a, false, b0, (short)0, acc[t + 0], false, false);
      if (t + 4 < TILES_N) b0 = GVQ_LOADB(t + 4);
      acc[t + 1] = __builtin_amdgcn_wmma_f32_16x16x32_bf16(
          false, a, false, b1, (short)0, acc[t + 1], false, false);
      if (t + 5 < TILES_N) b1 = GVQ_LOADB(t + 5);
      acc[t + 2] = __builtin_amdgcn_wmma_f32_16x16x32_bf16(
          false, a, false, b2, (short)0, acc[t + 2], false, false);
      if (t + 6 < TILES_N) b2 = GVQ_LOADB(t + 6);
      acc[t + 3] = __builtin_amdgcn_wmma_f32_16x16x32_bf16(
          false, a, false, b3, (short)0, acc[t + 3], false, false);
      if (t + 7 < TILES_N) b3 = GVQ_LOADB(t + 7);
    }
    #undef GVQ_LOADB

#if GVQ_SCHED
    // Force a software pipeline in the pre-RA scheduler: 3 B fragments of
    // DS reads (6 x ds_load_b128) hoisted ahead of the first WMMA, then
    // alternate 1 WMMA / 2 DS reads, then drain the last 3 WMMAs. This
    // keeps >=3 distinct B register sets live so no WMMA needs
    // s_wait_dscnt 0.   masks: 0x008 = MFMA/WMMA, 0x100 = DS read.
    __builtin_amdgcn_sched_group_barrier(0x100, 6, 0);
    #pragma unroll
    for (int t = 0; t < TILES_N - 3; ++t) {
      __builtin_amdgcn_sched_group_barrier(0x008, 1, 0);
      __builtin_amdgcn_sched_group_barrier(0x100, 2, 0);
    }
    __builtin_amdgcn_sched_group_barrier(0x008, 3, 0);
#endif

    gvq_stage_wait(wave);   // next chunk landed before anyone reads it
    __syncthreads();
  }

  // ---- Bs now dead: reuse LDS for stats ----
  for (int i = tid; i < GVDIM; i += 256) {
    smem.st.probs[i]  = 0.0f;
    smem.st.counts[i] = 0.0f;
  }
  __syncthreads();

  // ---- bias add (C/D layout: lane holds col t*16+ln, rows r + 8h) ----
  #pragma unroll
  for (int t = 0; t < TILES_N; ++t) {
    const float bc = bias[gw * VCODES + t * 16 + ln];
    #pragma unroll
    for (int r = 0; r < 8; ++r) acc[t][r] += bc;
  }

  // ---- per-row stats: softmax mean, hard argmax, gumbel argmax ----
  #pragma unroll
  for (int r = 0; r < 8; ++r) {
    const int nloc  = mw * 16 + r + 8 * h;          // block-local row 0..63
    const int nglob = blk * BLK_M + nloc;           // global row 0..16383

    // softmax over 320 cols (16 lanes x 20 tiles, per half-wave)
    float mx = -3.4e38f;
    #pragma unroll
    for (int t = 0; t < TILES_N; ++t) mx = fmaxf(mx, acc[t][r]);
    #pragma unroll
    for (int msk = 1; msk < 16; msk <<= 1) mx = fmaxf(mx, __shfl_xor(mx, msk, 16));
    float ev[TILES_N]; float s = 0.0f;
    #pragma unroll
    for (int t = 0; t < TILES_N; ++t) { ev[t] = __expf(acc[t][r] - mx); s += ev[t]; }
    #pragma unroll
    for (int msk = 1; msk < 16; msk <<= 1) s += __shfl_xor(s, msk, 16);
    const float inv = 1.0f / s;
    #pragma unroll
    for (int t = 0; t < TILES_N; ++t)
      __hip_atomic_fetch_add(&smem.st.probs[gw * VCODES + t * 16 + ln], ev[t] * inv,
                             __ATOMIC_RELAXED, __HIP_MEMORY_SCOPE_WORKGROUP);

    // hard argmax of raw logits (first-max tie break = lowest index)
    float bv = acc[0][r]; int bi = ln;
    #pragma unroll
    for (int t = 1; t < TILES_N; ++t) {
      const float v = acc[t][r];
      if (v > bv) { bv = v; bi = t * 16 + ln; }
    }
    #pragma unroll
    for (int msk = 1; msk < 16; msk <<= 1) {
      const float ov = __shfl_xor(bv, msk, 16);
      const int   oi = __shfl_xor(bi, msk, 16);
      if (ov > bv || (ov == bv && oi < bi)) { bv = ov; bi = oi; }
    }
    if (ln == 0)
      __hip_atomic_fetch_add(&smem.st.counts[gw * VCODES + bi], 1.0f,
                             __ATOMIC_RELAXED, __HIP_MEMORY_SCOPE_WORKGROUP);

    // gumbel hard index: argmax(l + gumbel) (temperature is monotone)
    const float* gr = gn + ((size_t)nglob * NGROUPS + gw) * VCODES;
    float gv = acc[0][r] + gr[ln]; int gi = ln;
    #pragma unroll
    for (int t = 1; t < TILES_N; ++t) {
      const float v = acc[t][r] + gr[t * 16 + ln];
      if (v > gv) { gv = v; gi = t * 16 + ln; }
    }
    #pragma unroll
    for (int msk = 1; msk < 16; msk <<= 1) {
      const float ov = __shfl_xor(gv, msk, 16);
      const int   oi = __shfl_xor(gi, msk, 16);
      if (ov > gv || (ov == gv && oi < gi)) { gv = ov; gi = oi; }
    }
    if (ln == 0) smem.st.idx[gw * BLK_M + nloc] = gi;
  }

  __syncthreads();

  // flush block-local stats (one atomic per entry per block)
  for (int i = tid; i < GVDIM; i += 256) {
    __hip_atomic_fetch_add(&g_probs[i],  smem.st.probs[i],
                           __ATOMIC_RELAXED, __HIP_MEMORY_SCOPE_AGENT);
    __hip_atomic_fetch_add(&g_counts[i], smem.st.counts[i],
                           __ATOMIC_RELAXED, __HIP_MEMORY_SCOPE_AGENT);
  }

  // q gather: straight-through forward value = hard one-hot @ codebook
  for (int p = wave; p < NGROUPS * BLK_M; p += 8) {
    const int nloc = p >> 1;
    const int g    = p & 1;
    const int idx  = smem.st.idx[g * BLK_M + nloc];
    const float4* src = (const float4*)(codebook + (size_t)(g * VCODES + idx) * D_DIM);
    float4* dst = (float4*)(out + (size_t)(blk * BLK_M + nloc) * (NGROUPS * D_DIM) + g * D_DIM);
    dst[lane]      = src[lane];
    dst[lane + 32] = src[lane + 32];
  }
}

// ---------------------------------------------------------------------------
// Final scalars: code/prob perplexity + diversity loss.
// ---------------------------------------------------------------------------
__global__ void gvq_scalars_kernel(const float* __restrict__ g_probs,
                                   const float* __restrict__ g_counts,
                                   float* __restrict__ out3) {
  __shared__ float red[4][256];
  const int tid = threadIdx.x;
  float cH0 = 0.f, cH1 = 0.f, pH0 = 0.f, pH1 = 0.f;
  for (int e = tid; e < GVDIM; e += 256) {
    const float hp = g_counts[e] * (1.0f / (float)NROWS);
    const float pp = g_probs[e]  * (1.0f / (float)NROWS);
    const float ch = hp * __logf(hp + 1e-7f);
    const float ph = pp * __logf(pp + 1e-7f);
    if (e < VCODES) { cH0 += ch; pH0 += ph; } else { cH1 += ch; pH1 += ph; }
  }
  red[0][tid] = cH0; red[1][tid] = cH1; red[2][tid] = pH0; red[3][tid] = pH1;
  __syncthreads();
  for (int s = 128; s > 0; s >>= 1) {
    if (tid < s) {
      red[0][tid] += red[0][tid + s];
      red[1][tid] += red[1][tid + s];
      red[2][tid] += red[2][tid + s];
      red[3][tid] += red[3][tid + s];
    }
    __syncthreads();
  }
  if (tid == 0) {
    const float code_ppl = __expf(-red[0][0]) + __expf(-red[1][0]);
    const float prob_ppl = __expf(-red[2][0]) + __expf(-red[3][0]);
    out3[0] = code_ppl;
    out3[1] = prob_ppl;
    out3[2] = ((float)GVDIM - prob_ppl) / (float)GVDIM;
  }
}

// ---------------------------------------------------------------------------
extern "C" void kernel_launch(void* const* d_in, const int* in_sizes, int n_in,
                              void* d_out, int out_size, void* d_ws, size_t ws_size,
                              hipStream_t stream) {
  const float* x  = (const float*)d_in[0];   // [8,2048,1024]
  const float* W  = (const float*)d_in[1];   // [640,1024]
  const float* b  = (const float*)d_in[2];   // [640]
  const float* cb = (const float*)d_in[3];   // [1,640,256]
  const float* gn = (const float*)d_in[4];   // [32768,320]
  float* out = (float*)d_out;                // q (8,388,608) + 3 scalars

  // workspace: [g_probs 640 f32][g_counts 640 f32][pad][W bf16 640x1024]
  float* g_probs  = (float*)d_ws;
  float* g_counts = g_probs + GVDIM;
  unsigned short* Wb = (unsigned short*)((char*)d_ws + 8192);

  gvq_zero_kernel<<<(2 * GVDIM + 255) / 256, 256, 0, stream>>>(g_probs, 2 * GVDIM);
  gvq_wconv_kernel<<<(GVDIM * C_DIM + 255) / 256, 256, 0, stream>>>(W, Wb, GVDIM * C_DIM);
  gvq_fused_kernel<<<NROWS / BLK_M, 256, 0, stream>>>(x, Wb, b, cb, gn, out,
                                                      g_probs, g_counts);
  gvq_scalars_kernel<<<1, 256, 0, stream>>>(g_probs, g_counts,
                                            out + (size_t)NROWS * NGROUPS * D_DIM);
}
